// Int4LinearDequant_44092134261132
// MI455X (gfx1250) — matrix operand order repaired and checked by reading, not compiled, not run on hardware
//
#include <hip/hip_runtime.h>

// Int4-weight linear: out[M,N] = x[M,K] @ (q[N,K]*scale[N])^T + bias[N]
//   M = B*S = 8192, K = 4096, N = 11008
// Strategy: acc = x_bf16 @ q_bf16 (int4 exact in bf16) via v_wmma_f32_16x16x32_bf16,
// then out = acc*scale + bias.  Memory-bound (~585 MB @ 23.3 TB/s ~ 25us),
// so the kernel is structured around one pass over x / qweight through LDS.

#define IN_F    4096
#define OUT_F   11008
#define PACKEDW 2048   // int32 words per output row; each word carries ONE byte = 2 nibbles
#define MROWS   8192   // B*S
#define BM      128
#define BN      128
#define BK      32

typedef __bf16 bf16_t;
typedef __attribute__((ext_vector_type(16))) __bf16 v16bf;
typedef __attribute__((ext_vector_type(8)))  __bf16 v8bf;
typedef __attribute__((ext_vector_type(8)))  float  v8f;

union V16u { v16bf v; v8bf h[2]; };

// round-to-nearest-even fp32 -> bf16 bit pattern (returned in low 16 bits)
__device__ __forceinline__ unsigned f2bf(float f) {
    union { float f; unsigned u; } v; v.f = f;
    return ((v.u + 0x7FFFu + ((v.u >> 16) & 1u)) >> 16) & 0xFFFFu;
}

__global__ __launch_bounds__(256)
void int4_linear_wmma_kernel(const float* __restrict__ x,
                             const int*   __restrict__ qw,
                             const float* __restrict__ scale,
                             const float* __restrict__ bias,
                             float*       __restrict__ out) {
    const int tid  = threadIdx.x;
    const int lane = tid & 31;          // wave32
    const int wave = tid >> 5;          // 0..7
    const int m0   = blockIdx.y * BM;
    const int n0   = blockIdx.x * BN;
    const int mW   = (wave >> 2) * 64;  // wave M offset within block tile
    const int nW   = (wave & 3) * 32;   // wave N offset within block tile
    const int lrow = lane & 15;
    const int half = lane >> 4;         // 0: lanes 0-15, 1: lanes 16-31

    // LDS tiles: A = x (BM x BK bf16), B = q (BN x BK bf16), 16 KB total
    __shared__ __align__(16) unsigned short Abuf[BM * BK];
    __shared__ __align__(16) unsigned short Bbuf[BN * BK];

    const v8f zero = {0.f, 0.f, 0.f, 0.f, 0.f, 0.f, 0.f, 0.f};
    v8f acc[4][2];
    #pragma unroll
    for (int i = 0; i < 4; ++i)
        #pragma unroll
        for (int j = 0; j < 2; ++j)
            acc[i][j] = zero;

    for (int kk = 0; kk < IN_F; kk += BK) {
        // ---- cooperative load: x tile (128x32 fp32 -> bf16), 4 float4 per thread ----
        #pragma unroll
        for (int it = 0; it < 4; ++it) {
            const int f   = tid + it * 256;       // float4 index 0..1023
            const int row = f >> 3;               // 8 float4 per row of 32
            const int kq  = f & 7;
            const float4 v = *reinterpret_cast<const float4*>(
                x + (size_t)(m0 + row) * IN_F + kk + kq * 4);
            unsigned* Ar = reinterpret_cast<unsigned*>(&Abuf[row * BK]);
            Ar[kq * 2 + 0] = f2bf(v.x) | (f2bf(v.y) << 16);
            Ar[kq * 2 + 1] = f2bf(v.z) | (f2bf(v.w) << 16);
        }
        // ---- cooperative load: qweight tile (128 rows x 16 words), decode nibbles ----
        // word j of a k-block -> k=2j (low nibble), k=2j+1 (high nibble); values-8 exact in bf16
        #pragma unroll
        for (int it = 0; it < 2; ++it) {
            const int g   = tid + it * 256;       // int4 index 0..511
            const int row = g >> 2;               // 4 int4 per row of 16 words
            const int wq  = g & 3;
            const int4 wv = *reinterpret_cast<const int4*>(
                qw + (size_t)(n0 + row) * PACKEDW + (kk >> 1) + wq * 4);
            unsigned* Br = reinterpret_cast<unsigned*>(&Bbuf[row * BK]);
            const int wsv[4] = {wv.x, wv.y, wv.z, wv.w};
            #pragma unroll
            for (int j = 0; j < 4; ++j) {
                const int b  = wsv[j] & 0xFF;
                const float lo = (float)((b & 15) - 8);
                const float hi = (float)(((b >> 4) & 15) - 8);
                Br[wq * 4 + j] = f2bf(lo) | (f2bf(hi) << 16);
            }
        }
        // prefetch next K-tile into cache while waiting at the barrier
        if (kk + BK < IN_F) {
            __builtin_prefetch(x  + (size_t)(m0 + (tid >> 1)) * IN_F    + kk + BK,        0, 3);
            __builtin_prefetch(qw + (size_t)(n0 + (tid >> 1)) * PACKEDW + ((kk + BK) >> 1), 0, 3);
        }
        __syncthreads();

        const bf16_t* Ab = reinterpret_cast<const bf16_t*>(Abuf);
        const bf16_t* Bb = reinterpret_cast<const bf16_t*>(Bbuf);

        // A fragment (16x32 bf16): lane L (0-15) row M=L holds K {0..7,16..23};
        // lanes 16-31 row M=L-16 hold K {8..15,24..31}  -> two ds_load_b128 each
        v16bf af[4];
        #pragma unroll
        for (int i = 0; i < 4; ++i) {
            const bf16_t* p = Ab + (mW + i * 16 + lrow) * BK + half * 8;
            V16u u;
            u.h[0] = *reinterpret_cast<const v8bf*>(p);        // K base .. +7
            u.h[1] = *reinterpret_cast<const v8bf*>(p + 16);   // K base+16 .. +23
            af[i] = u.v;
        }
        // B fragment (32x16 bf16): lane -> N = lane%16; lanes 0-15 K {0..7,16..23}, lanes 16-31 K {8..15,24..31}
        v16bf bfr[2];
        #pragma unroll
        for (int j = 0; j < 2; ++j) {
            const bf16_t* p = Bb + (nW + j * 16 + lrow) * BK + half * 8;
            V16u u;
            u.h[0] = *reinterpret_cast<const v8bf*>(p);
            u.h[1] = *reinterpret_cast<const v8bf*>(p + 16);
            bfr[j] = u.v;
        }

        #pragma unroll
        for (int i = 0; i < 4; ++i)
            #pragma unroll
            for (int j = 0; j < 2; ++j)
                acc[i][j] = __builtin_amdgcn_wmma_f32_16x16x32_bf16(
                    false, af[i], false, bfr[j], (short)0, acc[i][j], false, false);

        __syncthreads();
    }

    // ---- epilogue: out = acc*scale[col] + bias[col] ----
    // C/D layout: VGPR r -> M = r (lanes 0-15) or 8+r (lanes 16-31); N = lane%16
    #pragma unroll
    for (int j = 0; j < 2; ++j) {
        const int gcol = n0 + nW + j * 16 + lrow;
        const float s  = scale[gcol];
        const float bb = bias[gcol];
        #pragma unroll
        for (int i = 0; i < 4; ++i) {
            const int growb = m0 + mW + i * 16 + half * 8;
            #pragma unroll
            for (int r = 0; r < 8; ++r) {
                out[(size_t)(growb + r) * OUT_F + gcol] = acc[i][j][r] * s + bb;
            }
        }
    }
}

extern "C" void kernel_launch(void* const* d_in, const int* in_sizes, int n_in,
                              void* d_out, int out_size, void* d_ws, size_t ws_size,
                              hipStream_t stream) {
    const float* x     = (const float*)d_in[0];
    const int*   qw    = (const int*)d_in[1];
    const float* scale = (const float*)d_in[2];
    const float* bias  = (const float*)d_in[3];
    float*       out   = (float*)d_out;

    dim3 grid(OUT_F / BN, MROWS / BM);   // (86, 64)
    dim3 block(256);
    int4_linear_wmma_kernel<<<grid, block, 0, stream>>>(x, qw, scale, bias, out);
}
